// CrossTransformer_score2_37520834298094
// MI455X (gfx1250) — matrix operand order repaired and checked by reading, not compiled, not run on hardware
//
#include <hip/hip_runtime.h>
#include <hip/hip_bf16.h>
#include <stdint.h>

// ---------- CDNA5 WMMA types ----------
typedef __attribute__((ext_vector_type(16))) __bf16       v16bf;
typedef __attribute__((ext_vector_type(8)))  float        v8f;
typedef __attribute__((ext_vector_type(4)))  unsigned int v4u;

union Frag16 {
  v16bf v;
  unsigned short s[16];
  v4u q[2];
};

__device__ __forceinline__ unsigned short f2bf(float f) {
  union { float f; unsigned u; } x; x.f = f;
  unsigned u = x.u;
  u += 0x7fffu + ((u >> 16) & 1u);   // round-to-nearest-even
  return (unsigned short)(u >> 16);
}

__device__ __forceinline__ unsigned pack2bf(float a, float b) {
  return (unsigned)f2bf(a) | ((unsigned)f2bf(b) << 16);
}

#define NB     8
#define NWAY   5
#define KSHOT  5
#define HW     196
#define C_IN   512
#define DKEY   128
#define NSAMP  208            // 8 query + 200 supports
#define NSUP   200
#define NBN    40             // NB*NWAY
#define NKEYS  980            // KSHOT*HW
#define KPAD   992            // keys padded to 62*16
#define QSCALE 0.08838834764831845f   // 128^-0.5

// ---------------- kernel 1: channel mean of supports ----------------
__global__ void k_sup_avg(const float* __restrict__ sup, float* __restrict__ avg) {
  const int c = blockIdx.x;
  float s = 0.f;
  for (int i = threadIdx.x; i < NSUP * HW; i += 256) {
    int sm = i / HW, p = i - sm * HW;
    s += sup[((size_t)sm * C_IN + c) * HW + p];
  }
  __shared__ float red[256];
  red[threadIdx.x] = s;
  __syncthreads();
  for (int st = 128; st > 0; st >>= 1) {
    if (threadIdx.x < st) red[threadIdx.x] += red[threadIdx.x + st];
    __syncthreads();
  }
  if (threadIdx.x == 0) avg[c] = red[0] * (1.0f / (float)(NSUP * HW));
}

// ---------------- kernel 2: sigmoid weight table (64 ch per pixel) ----------------
__global__ void k_weight(const float* __restrict__ q, const float* __restrict__ avg,
                         const float* __restrict__ ws1, const float* __restrict__ ws2,
                         float* __restrict__ wt) {
  const int bp = blockIdx.x;
  const int b = bp / HW, p = bp - b * HW;
  const int j = threadIdx.x;               // 64 threads
  const float* w1 = ws1 + (size_t)j * (2 * C_IN);
  const float* qb = q + (size_t)b * C_IN * HW;
  float acc = 0.f;
  for (int c = 0; c < C_IN; ++c) acc += w1[c] * qb[(size_t)c * HW + p];
  for (int c = 0; c < C_IN; ++c) acc += w1[C_IN + c] * avg[c];
  __shared__ float h1[64];
  h1[j] = acc;
  __syncthreads();
  const float* w2 = ws2 + (size_t)j * 64;
  float a2 = 0.f;
  for (int i = 0; i < 64; ++i) a2 += w2[i] * h1[i];
  wt[((size_t)b * 64 + j) * HW + p] = 1.0f / (1.0f + __expf(-a2));
}

// ---------------- kernel 3: weights -> bf16 ----------------
__global__ void k_cvt_w(const float* __restrict__ wqk, const float* __restrict__ wv,
                        unsigned short* __restrict__ wbf) {
  const int i = blockIdx.x * 256 + threadIdx.x;   // 0 .. 131071
  const float v = (i < DKEY * C_IN) ? wqk[i] : wv[i - DKEY * C_IN];
  wbf[i] = f2bf(v);
}

// ---------------- kernel 3b: zero the key-pad slices of kpad/vpad ----------------
__global__ void k_zero_pad(unsigned int* __restrict__ kpad32, unsigned int* __restrict__ vpad32) {
  const int i = blockIdx.x * 256 + threadIdx.x;   // 0 .. 30719
  if (i < NBN * 12 * 64) {        // kpad: 40 bn x 12 pad-keys x 64 u32 (128 bf16)
    int bn = i / (12 * 64);
    int rem = i - bn * 12 * 64;
    int k = rem >> 6, c = rem & 63;
    kpad32[((size_t)bn * KPAD + NKEYS + k) * 64 + c] = 0;
  }
  if (i < NBN * DKEY * 6) {       // vpad: 40 bn x 128 ch x 6 u32 (12 bf16)
    int bn = i / (DKEY * 6);
    int rem = i - bn * (DKEY * 6);
    int c = rem / 6, j = rem - c * 6;
    vpad32[((size_t)bn * DKEY + c) * (KPAD / 2) + (NKEYS / 2) + j] = 0;
  }
}

// ---------------- kernel 4: projections via bf16 WMMA ----------------
// grid: (sample 0..207, ntile 0..12, wsel 0..1), block 256 (8 waves = 8 M-tiles)
// wsel 0 (w_qk): query -> qt[b][px][128];   supports -> kpad[bn][key][128]
// wsel 1 (w_v):  query -> qvf f32 [b][128][px]; supports -> vpad[bn][128][key]
__global__ void k_project(const float* __restrict__ qrep, const float* __restrict__ srep,
                          const unsigned short* __restrict__ wbf,
                          unsigned short* __restrict__ qt,
                          unsigned short* __restrict__ kpad,
                          unsigned short* __restrict__ vpad,
                          float* __restrict__ qvf) {
  const int sample = blockIdx.x;
  const int nbase = blockIdx.y * 16;
  const int wsel = blockIdx.z;
  const int tid = threadIdx.x;
  const int lane = tid & 31;
  const int wave = tid >> 5;           // M-tile
  const int half = lane >> 4;
  const int l16  = lane & 15;
  const int mbase = wave * 16;

  const float* x = (sample < NB) ? (qrep + (size_t)sample * C_IN * HW)
                                 : (srep + (size_t)(sample - NB) * C_IN * HW);
  const unsigned short* wmat = wbf + (size_t)wsel * DKEY * C_IN;

  // double-buffered B tile: [buf][px][ch], 2 KB
  __shared__ __attribute__((aligned(16))) unsigned short Bt[2][16 * 32];

  // fixed staging assignment: thread -> (pixel pxl, channel pair ch2)
  const int pxl = tid & 15;
  const int ch2 = (tid >> 4) << 1;                 // 0,2,...,30
  int pxc = nbase + pxl; if (pxc > HW - 1) pxc = HW - 1;  // clamp; masked at store
  const float* xp = x + (size_t)ch2 * HW + pxc;
  unsigned short* const bslot0 = &Bt[0][pxl * 32 + ch2];
  unsigned short* const bslot1 = &Bt[1][pxl * 32 + ch2];

  // prologue: stage k-step 0 into buffer 0
  *(unsigned*)bslot0 = pack2bf(xp[0], xp[HW]);
  __syncthreads();

  v8f acc = {};
  for (int step = 0; step < 16; ++step) {
    const int kb = step * 32;
    const int cur = step & 1;
    // prefetch next tile into the other buffer (overlaps WMMA below)
    if (step < 15) {
      const float* xn = xp + (size_t)(kb + 32) * HW;
      *(unsigned*)(cur ? bslot0 : bslot1) = pack2bf(xn[0], xn[HW]);
    }
    Frag16 a, bfr;
    {  // A: 16x32 bf16, lane = row, K = kb + half*16 + {0..15} contiguous
      const unsigned short* ap = wmat + (size_t)(mbase + l16) * C_IN + kb + half * 16;
      a.q[0] = *(const v4u*)(ap);
      a.q[1] = *(const v4u*)(ap + 8);
    }
    {  // B: 32x16 bf16, lane = col, K = half*16 + {0..15} contiguous
      const unsigned short* bp = &Bt[cur][l16 * 32 + half * 16];
      bfr.q[0] = *(const v4u*)(bp);
      bfr.q[1] = *(const v4u*)(bp + 8);
    }
    acc = __builtin_amdgcn_wmma_f32_16x16x32_bf16(false, a.v, false, bfr.v,
                                                  (short)0, acc, false, false);
    __syncthreads();   // reads of cur done; next step may overwrite it
  }

  const int px = nbase + l16;
  if (px < HW) {
    // D elem r -> channel mbase + half*8 + r, col = px
    const int m0 = mbase + half * 8;
    if (sample < NB) {
      if (wsel == 0) {
        unsigned short* dst = qt + ((size_t)sample * HW + px) * DKEY + m0;
        for (int r = 0; r < 8; ++r) dst[r] = f2bf(acc[r]);
      } else {
        for (int r = 0; r < 8; ++r)
          qvf[((size_t)sample * DKEY + m0 + r) * HW + px] = acc[r];
      }
    } else {
      const int sp = sample - NB;
      const int bn = sp / KSHOT, shot = sp - bn * KSHOT;
      const int key = shot * HW + px;
      if (wsel == 0) {
        unsigned short* dst = kpad + ((size_t)bn * KPAD + key) * DKEY + m0;
        for (int r = 0; r < 8; ++r) dst[r] = f2bf(acc[r]);
      } else {
        for (int r = 0; r < 8; ++r)
          vpad[((size_t)bn * DKEY + m0 + r) * KPAD + key] = f2bf(acc[r]);
      }
    }
  }
}

// ---------------- kernel 5: flash attention + weighted L2 partial ----------------
// grid (b, n, mtile 0..12), 1 wave (32 threads) per block.
__global__ void k_attn(const unsigned short* __restrict__ qt,    // [8][196][128]  bf16
                       const unsigned short* __restrict__ kpad,  // [40][992][128] bf16
                       const unsigned short* __restrict__ vpad,  // [40][128][992] bf16
                       const float* __restrict__ qvf,            // [8][128][196]
                       const float* __restrict__ wt,             // [8][64][196]
                       float* __restrict__ partial) {
  const int b = blockIdx.x;
  const int ni = blockIdx.y;
  const int mtile = blockIdx.z;
  const int lane = threadIdx.x;
  const int half = lane >> 4, l16 = lane & 15;
  const int mbase = mtile * 16;
  const int bn = b * NWAY + ni;

  __shared__ __attribute__((aligned(16))) unsigned short Plds[16 * 32];

  // Q A-fragments (row = query pixel, K = channel); clamp pad rows
  Frag16 aq[4];
  {
    int qp = mbase + l16; if (qp > HW - 1) qp = HW - 1;
    const unsigned short* qrow = qt + ((size_t)b * HW + qp) * DKEY;
#pragma unroll
    for (int kk = 0; kk < 4; ++kk) {
      const unsigned short* p = qrow + kk * 32 + half * 16;
      aq[kk].q[0] = *(const v4u*)(p);
      aq[kk].q[1] = *(const v4u*)(p + 8);
    }
  }

  v8f o[8] = {};                       // out: 16 qpix x 128 ch (8 N-tiles)
  float mrow[8], srow[8];
  for (int r = 0; r < 8; ++r) { mrow[r] = -1e30f; srow[r] = 0.f; }

  for (int kb = 0; kb < KPAD; kb += 32) {         // 31 key-blocks of 32 (980 valid)
    float S[2][8];
#pragma unroll
    for (int sub = 0; sub < 2; ++sub) {
      const int l = kb + sub * 16 + l16;           // < 992, always in-bounds in kpad
      const bool valid = (l < NKEYS);
      const unsigned short* krow = kpad + ((size_t)bn * KPAD + l) * DKEY;
      v8f acc = {};
#pragma unroll
      for (int kk = 0; kk < 4; ++kk) {
        Frag16 bf;
        const unsigned short* p = krow + kk * 32 + half * 16;
        bf.q[0] = *(const v4u*)(p);
        bf.q[1] = *(const v4u*)(p + 8);
        acc = __builtin_amdgcn_wmma_f32_16x16x32_bf16(false, aq[kk].v, false, bf.v,
                                                      (short)0, acc, false, false);
      }
#pragma unroll
      for (int r = 0; r < 8; ++r) S[sub][r] = valid ? acc[r] * QSCALE : -1e30f;
    }

    // online softmax over the 32 columns (rows live as r + 8*half; cols in l16 lanes)
    float cf[8];
#pragma unroll
    for (int r = 0; r < 8; ++r) {
      float tm = fmaxf(S[0][r], S[1][r]);
      for (int m = 1; m < 16; m <<= 1) tm = fmaxf(tm, __shfl_xor(tm, m, 32));
      float mn = fmaxf(mrow[r], tm);
      cf[r] = __expf(mrow[r] - mn);
      mrow[r] = mn;
      S[0][r] = __expf(S[0][r] - mn);
      S[1][r] = __expf(S[1][r] - mn);
      float rs = S[0][r] + S[1][r];
      for (int m = 1; m < 16; m <<= 1) rs += __shfl_xor(rs, m, 32);
      srow[r] = srow[r] * cf[r] + rs;
    }
#pragma unroll
    for (int ct = 0; ct < 8; ++ct)
#pragma unroll
      for (int r = 0; r < 8; ++r) o[ct][r] *= cf[r];

    // relayout P (D-layout) -> A-fragment layout through LDS
    __syncthreads();
#pragma unroll
    for (int r = 0; r < 8; ++r) {
      const int row = r + 8 * half;
      Plds[row * 32 + l16]      = f2bf(S[0][r]);
      Plds[row * 32 + 16 + l16] = f2bf(S[1][r]);
    }
    __syncthreads();
    Frag16 pf;
    {
      const unsigned short* p = &Plds[l16 * 32 + half * 16];
      pf.q[0] = *(const v4u*)(p);
      pf.q[1] = *(const v4u*)(p + 8);
    }

    // P (16x32) x V (32 keys x 128 ch), 8 channel N-tiles; V contiguous in vpad
#pragma unroll
    for (int ct = 0; ct < 8; ++ct) {
      Frag16 vf;
      const int c = ct * 16 + l16;
      const unsigned short* p = vpad + ((size_t)bn * DKEY + c) * KPAD + kb + half * 16;
      vf.q[0] = *(const v4u*)(p);
      vf.q[1] = *(const v4u*)(p + 8);
      o[ct] = __builtin_amdgcn_wmma_f32_16x16x32_bf16(false, pf.v, false, vf.v,
                                                      (short)0, o[ct], false, false);
    }
  }

  // normalize + weighted L2 partial for this M-tile
  float part = 0.f;
#pragma unroll
  for (int ct = 0; ct < 8; ++ct) {
    const int c = ct * 16 + l16;
#pragma unroll
    for (int r = 0; r < 8; ++r) {
      const int row = r + 8 * half;
      const int qp = mbase + row;
      if (qp < HW) {
        const float ov = o[ct][r] / srow[r];
        const float d = qvf[((size_t)b * DKEY + c) * HW + qp] - ov;
        const float w = wt[((size_t)b * 64 + (c & 63)) * HW + qp];
        part += w * d * d;
      }
    }
  }
  for (int m = 1; m < 32; m <<= 1) part += __shfl_xor(part, m, 32);
  if (lane == 0) partial[((size_t)bn) * 13 + mtile] = part;
}

// ---------------- kernel 6: deterministic final sum ----------------
__global__ void k_final(const float* __restrict__ partial, float* __restrict__ out) {
  const int t = threadIdx.x;
  if (t < NBN) {
    float s = 0.f;
    for (int i = 0; i < 13; ++i) s += partial[(size_t)t * 13 + i];
    out[t] = -s / (float)HW;
  }
}

// ---------------- launcher ----------------
extern "C" void kernel_launch(void* const* d_in, const int* in_sizes, int n_in,
                              void* d_out, int out_size, void* d_ws, size_t ws_size,
                              hipStream_t stream) {
  (void)in_sizes; (void)n_in; (void)out_size; (void)ws_size;
  const float* qrep = (const float*)d_in[0];   // [8,512,14,14]
  const float* srep = (const float*)d_in[1];   // [200,512,14,14]
  const float* wqk  = (const float*)d_in[2];   // [128,512]
  const float* wv   = (const float*)d_in[3];   // [128,512]
  const float* ws1  = (const float*)d_in[4];   // [64,1024]
  const float* ws2  = (const float*)d_in[5];   // [64,64]
  float* out = (float*)d_out;                  // [8,5]

  // workspace layout (~22.2 MB total)
  char* ws = (char*)d_ws;
  float*          avg  = (float*)(ws + 0);                 //   512 f32
  float*          wtb  = (float*)(ws + 4096);              // 8*64*196 f32   (401408 B)
  float*          qvf  = (float*)(ws + 405504);            // 8*128*196 f32  (802816 B)
  float*          part = (float*)(ws + 1208320);           //   520 f32
  unsigned short* wbf  = (unsigned short*)(ws + 1212416);  // 2*128*512  bf16 (262144 B)
  unsigned short* qt   = (unsigned short*)(ws + 1474560);  // 8*196*128  bf16 (401408 B)
  unsigned short* kpad = (unsigned short*)(ws + 1875968);  // 40*992*128 bf16 (10158080 B)
  unsigned short* vpad = (unsigned short*)(ws + 12034048); // 40*128*992 bf16 (10158080 B)

  k_sup_avg<<<dim3(C_IN), dim3(256), 0, stream>>>(srep, avg);
  k_weight<<<dim3(NB * HW), dim3(64), 0, stream>>>(qrep, avg, ws1, ws2, wtb);
  k_cvt_w<<<dim3(512), dim3(256), 0, stream>>>(wqk, wv, wbf);
  k_zero_pad<<<dim3(120), dim3(256), 0, stream>>>((unsigned int*)kpad, (unsigned int*)vpad);
  k_project<<<dim3(NSAMP, 13, 2), dim3(256), 0, stream>>>(qrep, srep, wbf, qt, kpad, vpad, qvf);
  k_attn<<<dim3(NB, NWAY, 13), dim3(32), 0, stream>>>(qt, kpad, vpad, qvf, wtb, part);
  k_final<<<dim3(1), dim3(64), 0, stream>>>(part, out);
}
